// GNNModel_78099685311022
// MI455X (gfx1250) — matrix operand order repaired and checked by reading, not compiled, not run on hardware
//
#include <hip/hip_runtime.h>
#include <hip/hip_bf16.h>

typedef __attribute__((ext_vector_type(16))) _Float16 v16h;
typedef __attribute__((ext_vector_type(8)))  float    v8f;

#define GDIM 64
#define SLOPE 0.01f

// ---------------------------------------------------------------- utilities
__global__ void __launch_bounds__(256) k_fill(float* __restrict__ p, float v, int n) {
    int i = blockIdx.x * 256 + threadIdx.x;
    if (i < n) p[i] = v;
}

// deg[dst] += w  (deg pre-filled with 1.0 for the self-loop)
__global__ void __launch_bounds__(256) k_deg(const int* __restrict__ dst,
                                             const float* __restrict__ w,
                                             float* __restrict__ deg, int E) {
    int e = blockIdx.x * 256 + threadIdx.x;
    if (e < E) atomicAdd(&deg[dst[e]], w[e]);
}

// in-place deg -> dinv
__global__ void __launch_bounds__(256) k_dinv(float* __restrict__ deg, int n) {
    int i = blockIdx.x * 256 + threadIdx.x;
    if (i < n) {
        float d = deg[i];
        deg[i] = (d > 0.f) ? rsqrtf(fmaxf(d, 1e-12f)) : 0.f;
    }
}

__global__ void __launch_bounds__(256) k_norm(const int* __restrict__ src,
                                              const int* __restrict__ dst,
                                              const float* __restrict__ w,
                                              const float* __restrict__ dinv,
                                              float* __restrict__ norm, int E) {
    int e = blockIdx.x * 256 + threadIdx.x;
    if (e < E) norm[e] = dinv[src[e]] * w[e] * dinv[dst[e]];
}

// ------------------------------------------------- WMMA GEMM: XW = X @ W
// X: [nrows, 64] f32, W: [64, 64] f32 (k-major), XW: [nrows, 64] f32.
// One wave per 16-row tile; f16 inputs, f32 accumulate. All intra-tile
// load/store offsets are compile-time constants folded into VMEM immediates.
__global__ void __launch_bounds__(256) k_gemm_wmma(const float* __restrict__ X,
                                                   const float* __restrict__ W,
                                                   float* __restrict__ XW, int nrows) {
    __shared__ _Float16 Wh[GDIM * GDIM];          // Wh[k*64 + n]
    int tid = threadIdx.x;
    for (int i = tid; i < GDIM * GDIM; i += 256) Wh[i] = (_Float16)W[i];
    __syncthreads();

    const int lane = tid & 31;
    const int h    = lane >> 4;                   // lane half (0/1)
    const int m    = lane & 15;                   // row/col within tile
    const int waveGlobal = blockIdx.x * 8 + (tid >> 5);
    const int numWaves   = gridDim.x * 8;

    // B fragments: element j of fragment (nt,ks) holds W[ks*32 + 8*h + j + (j>=8 ? 8 : 0)][nt*16 + m]
    v16h bfrag[4][2];
#pragma unroll
    for (int nt = 0; nt < 4; ++nt)
#pragma unroll
        for (int ks = 0; ks < 2; ++ks) {
            const int kb = ks * 32 + 8 * h;
            const int n  = nt * 16 + m;
            v16h b;
#pragma unroll
            for (int j = 0; j < 16; ++j) b[j] = Wh[(kb + j + (j >= 8 ? 8 : 0)) * GDIM + n];
            bfrag[nt][ks] = b;
        }

    const int ntiles = nrows >> 4;                // nrows is a multiple of 16
    for (int t = waveGlobal; t < ntiles; t += numWaves) {
        // A-row base with the lane-half offset folded in: remaining offsets
        // (ks*32 + j, +8 for the upper octet) are immediates.
        const float* xr = X + (size_t)(t * 16 + m) * GDIM + 8 * h;
        v8f acc[4] = {v8f{}, v8f{}, v8f{}, v8f{}};
#pragma unroll
        for (int ks = 0; ks < 2; ++ks) {
            v16h a;
#pragma unroll
            for (int j = 0; j < 8; ++j)  a[j] = (_Float16)xr[ks * 32 + j];
#pragma unroll
            for (int j = 8; j < 16; ++j) a[j] = (_Float16)xr[ks * 32 + 8 + j];
#pragma unroll
            for (int nt = 0; nt < 4; ++nt)
                acc[nt] = __builtin_amdgcn_wmma_f32_16x16x32_f16(
                    false, a, false, bfrag[nt][ks], (short)0, acc[nt], false, false);
        }
        // D layout: VGPR r -> M = r + 8*h, N = m. Single per-tile base; all 32
        // store offsets (r*64 + nt*16 floats) are immediates.
        float* base = XW + (size_t)t * (16 * GDIM) + (size_t)(8 * h) * GDIM + m;
#pragma unroll
        for (int r = 0; r < 8; ++r)
#pragma unroll
            for (int nt = 0; nt < 4; ++nt)
                base[r * GDIM + nt * 16] = acc[nt][r];
    }
}

// out[i][d] = b[d] + dinv[i]^2 * xw[i][d]   (self-loop term + bias, initializes out)
__global__ void __launch_bounds__(256) k_selfbias(const float* __restrict__ dinv,
                                                  const float* __restrict__ xw,
                                                  const float* __restrict__ b,
                                                  float* __restrict__ out, int n) {
    int idx = blockIdx.x * 256 + threadIdx.x;
    if (idx < n * GDIM) {
        int i = idx >> 6, d = idx & 63;
        float di = dinv[i];
        out[idx] = b[d] + di * di * xw[idx];
    }
}

// one wave per edge; lane owns float2 -> 256B coalesced gather + f32 atomic scatter
__global__ void __launch_bounds__(256) k_scatter(const int* __restrict__ src,
                                                 const int* __restrict__ dst,
                                                 const float* __restrict__ norm,
                                                 const float* __restrict__ xw,
                                                 float* __restrict__ out, int E) {
    int gtid = blockIdx.x * 256 + threadIdx.x;
    int e = gtid >> 5;
    if (e >= E) return;
    int lane = threadIdx.x & 31;
    float nv = norm[e];
    const float2 v = *(const float2*)(xw + (size_t)src[e] * GDIM + lane * 2);
    float* o = out + (size_t)dst[e] * GDIM + lane * 2;
    atomicAdd(o,     nv * v.x);
    atomicAdd(o + 1, nv * v.y);
}

__global__ void __launch_bounds__(256) k_leaky(float* __restrict__ x, int n) {
    int i = blockIdx.x * 256 + threadIdx.x;
    if (i < n) {
        float v = x[i];
        x[i] = (v > 0.f) ? v : SLOPE * v;
    }
}

// ------------------------------------------------------------------- head
__global__ void __launch_bounds__(256) k_head(const int* __restrict__ home,
                                              const int* __restrict__ away,
                                              const float* __restrict__ X,
                                              const float* __restrict__ l1W,   // [128,6]
                                              const float* __restrict__ l1b,   // [6]
                                              const float* __restrict__ l3W,   // [6,3]
                                              const float* __restrict__ l3b,   // [3]
                                              float* __restrict__ H3, int B) {
    int i = blockIdx.x * 256 + threadIdx.x;
    if (i >= B) return;
    const float* xh = X + (size_t)home[i] * GDIM;
    const float* xa = X + (size_t)away[i] * GDIM;
    float h[6];
#pragma unroll
    for (int j = 0; j < 6; ++j) h[j] = l1b[j];
    for (int k = 0; k < GDIM; ++k) {
        float v = xh[k];
#pragma unroll
        for (int j = 0; j < 6; ++j) h[j] += v * l1W[k * 6 + j];
    }
    for (int k = 0; k < GDIM; ++k) {
        float v = xa[k];
#pragma unroll
        for (int j = 0; j < 6; ++j) h[j] += v * l1W[(GDIM + k) * 6 + j];
    }
#pragma unroll
    for (int j = 0; j < 6; ++j) h[j] = (h[j] > 0.f) ? h[j] : SLOPE * h[j];
#pragma unroll
    for (int c = 0; c < 3; ++c) {
        float o = l3b[c];
#pragma unroll
        for (int j = 0; j < 6; ++j) o += h[j] * l3W[j * 3 + c];
        o = (o > 0.f) ? o : SLOPE * o;
        H3[i * 3 + c] = o;
    }
}

// single-block: per column c: red[c]=max_i, red[3+c]=log(sum_i exp(x-max))
__global__ void __launch_bounds__(256) k_lse(const float* __restrict__ H3,
                                             float* __restrict__ red, int B) {
    __shared__ float sm[256];
    int t = threadIdx.x;
    for (int c = 0; c < 3; ++c) {
        float mx = -1e30f;
        for (int i = t; i < B; i += 256) mx = fmaxf(mx, H3[i * 3 + c]);
        sm[t] = mx; __syncthreads();
        for (int s = 128; s > 0; s >>= 1) {
            if (t < s) sm[t] = fmaxf(sm[t], sm[t + s]);
            __syncthreads();
        }
        float m = sm[0]; __syncthreads();
        float sum = 0.f;
        for (int i = t; i < B; i += 256) sum += __expf(H3[i * 3 + c] - m);
        sm[t] = sum; __syncthreads();
        for (int s = 128; s > 0; s >>= 1) {
            if (t < s) sm[t] += sm[t + s];
            __syncthreads();
        }
        if (t == 0) { red[c] = m; red[3 + c] = logf(sm[0]); }
        __syncthreads();
    }
}

__global__ void __launch_bounds__(256) k_lsm_write(const float* __restrict__ H3,
                                                   const float* __restrict__ red,
                                                   float* __restrict__ out, int n3) {
    int i = blockIdx.x * 256 + threadIdx.x;
    if (i < n3) {
        int c = i % 3;
        out[i] = H3[i] - red[c] - red[3 + c];
    }
}

// ------------------------------------------------------------------- launch
extern "C" void kernel_launch(void* const* d_in, const int* in_sizes, int n_in,
                              void* d_out, int out_size, void* d_ws, size_t ws_size,
                              hipStream_t stream) {
    const int*   ei    = (const int*)d_in[0];      // [2,E]
    const float* ew    = (const float*)d_in[1];    // [E]
    const int*   home  = (const int*)d_in[2];      // [B]
    const int*   away  = (const int*)d_in[3];      // [B]
    const float* emb   = (const float*)d_in[4];    // [N,64]
    const float* W0    = (const float*)d_in[5];
    const float* b0    = (const float*)d_in[6];
    const float* W1    = (const float*)d_in[7];
    const float* b1    = (const float*)d_in[8];
    const float* W2    = (const float*)d_in[9];
    const float* b2    = (const float*)d_in[10];
    const float* l1W   = (const float*)d_in[11];   // [128,6]
    const float* l1b   = (const float*)d_in[12];
    const float* l3W   = (const float*)d_in[13];   // [6,3]
    const float* l3b   = (const float*)d_in[14];
    float* out = (float*)d_out;

    const int E = in_sizes[1];
    const int B = in_sizes[2];
    const int N = in_sizes[4] / GDIM;
    const int* src = ei;
    const int* dst = ei + E;

    // workspace carve-out
    char* ws = (char*)d_ws;
    size_t off = 0;
    auto carve = [&](size_t bytes) {
        void* p = ws + off;
        off = (off + bytes + 255) & ~(size_t)255;
        return p;
    };
    float* dinv = (float*)carve((size_t)N * 4);            // deg -> dinv in place
    float* norm = (float*)carve((size_t)E * 4);
    float* xw   = (float*)carve((size_t)N * GDIM * 4);
    float* xa   = (float*)carve((size_t)N * GDIM * 4);
    float* xb   = (float*)carve((size_t)N * GDIM * 4);
    float* h3   = (float*)carve((size_t)B * 3 * 4);
    float* red  = (float*)carve(8 * 4);

    const int TB = 256;
    auto nb = [&](long long n) { return (int)((n + TB - 1) / TB); };

    // ---- shared norm precompute (deg includes self-loop weight 1.0)
    k_fill<<<nb(N), TB, 0, stream>>>(dinv, 1.0f, N);
    k_deg <<<nb(E), TB, 0, stream>>>(dst, ew, dinv, E);
    k_dinv<<<nb(N), TB, 0, stream>>>(dinv, N);
    k_norm<<<nb(E), TB, 0, stream>>>(src, dst, ew, dinv, norm, E);

    // ---- 3 GCN layers
    const float* Ws[3] = {W0, W1, W2};
    const float* bs[3] = {b0, b1, b2};
    const float* xin[3] = {emb, xa, xb};
    float*       xout[3] = {xa, xb, xa};
    for (int l = 0; l < 3; ++l) {
        k_gemm_wmma<<<120, TB, 0, stream>>>(xin[l], Ws[l], xw, N);
        k_selfbias <<<nb((long long)N * GDIM), TB, 0, stream>>>(dinv, xw, bs[l], xout[l], N);
        k_scatter  <<<nb((long long)E * 32), TB, 0, stream>>>(src, dst, norm, xw, xout[l], E);
        k_leaky    <<<nb((long long)N * GDIM), TB, 0, stream>>>(xout[l], N * GDIM);
    }

    // ---- head + batch-axis log-softmax
    k_head<<<nb(B), TB, 0, stream>>>(home, away, xa, l1W, l1b, l3W, l3b, h3, B);
    k_lse <<<1, TB, 0, stream>>>(h3, red, B);
    k_lsm_write<<<nb((long long)B * 3), TB, 0, stream>>>(h3, red, out, B * 3);
}
